// EdgeStateInit_35691178230143
// MI455X (gfx1250) — matrix operand order repaired and checked by reading, not compiled, not run on hardware
//
#include <hip/hip_runtime.h>
#include <hip/hip_bf16.h>

typedef __attribute__((ext_vector_type(16))) __bf16          v16bf;
typedef __attribute__((ext_vector_type(8)))  float           v8f;
typedef __attribute__((ext_vector_type(4)))  unsigned short  us4;
typedef __attribute__((ext_vector_type(8)))  unsigned short  us8;
typedef __attribute__((ext_vector_type(16))) unsigned short  us16;

#define MBLK 64
#define SA_STRIDE 520   // 512 + 8 bf16 pad -> row stride 1040B, lanes land on distinct banks
#define SH_STRIDE 264   // 256 + 8 bf16 pad

static __device__ __forceinline__ unsigned short f2bf(float f) {
  // round-to-nearest-even f32 -> bf16 (NaN edge cases ignored)
  unsigned int u = __builtin_bit_cast(unsigned int, f);
  u += 0x7FFFu + ((u >> 16) & 1u);
  return (unsigned short)(u >> 16);
}

static __device__ __forceinline__ v16bf mk_frag(us8 lo, us8 hi) {
  us16 u = __builtin_shufflevector(lo, hi, 0,1,2,3,4,5,6,7,8,9,10,11,12,13,14,15);
  return __builtin_bit_cast(v16bf, u);
}

static __device__ __forceinline__ v16bf frag16(const unsigned short* p) {
  return __builtin_bit_cast(v16bf, *(const us16*)p);
}

// ---------------------------------------------------------------------------
// One-time weight pack: W1[256][513] f32 -> W1p[256][512] bf16 (col 512 stays
// f32, handled in epilogue); W2[256][256] f32 -> W2p bf16.
// ---------------------------------------------------------------------------
__global__ void pack_weights_kernel(const float* __restrict__ W1,
                                    const float* __restrict__ W2,
                                    unsigned short* __restrict__ W1p,
                                    unsigned short* __restrict__ W2p) {
  int idx = blockIdx.x * blockDim.x + threadIdx.x;
  if (idx < 256 * 512) {
    int c = idx >> 9, k = idx & 511;
    W1p[idx] = f2bf(W1[c * 513 + k]);
  } else if (idx < 256 * 512 + 256 * 256) {
    int j = idx - 256 * 512;
    W2p[j] = f2bf(W2[j]);
  }
}

// ---------------------------------------------------------------------------
// Fused gather + MLP(513->256, SiLU, 256->256) for 64 edges per workgroup.
// 8 waves; wave w owns output columns [32w, 32w+32) for all 64 edge rows:
// 4 M-tiles x 2 N-tiles of 16x16 f32 accumulators via v_wmma_f32_16x16x32_bf16.
// ---------------------------------------------------------------------------
__global__ __launch_bounds__(256, 2) void edge_mlp_kernel(
    const float* __restrict__ scalars,      // [N,256] f32
    const int*   __restrict__ edge_index,   // [2,E]
    const float* __restrict__ edge_len,     // [E]
    const float* __restrict__ W1,           // [256,513] f32 (only col 512 read)
    const float* __restrict__ b1,           // [256]
    const float* __restrict__ b2,           // [256]
    const unsigned short* __restrict__ W1p, // [256,512] bf16
    const unsigned short* __restrict__ W2p, // [256,256] bf16
    float* __restrict__ out,                // [E,256] f32
    int E) {
  __shared__ unsigned short sA[MBLK * SA_STRIDE];  // gathered msg_in, bf16
  __shared__ unsigned short sH[MBLK * SH_STRIDE];  // hidden activations, bf16
  __shared__ float sEl[MBLK];

  const int tid   = threadIdx.x;
  const int ebase = blockIdx.x * MBLK;

  // ---- stage A: gather sender|receiver features, convert to bf16 ----
  {
    int el = tid >> 2;                 // local edge 0..63
    int q  = tid & 3;                  // quarter: 0,1=sender halves; 2,3=receiver halves
    int eg = ebase + el; if (eg >= E) eg = E - 1;
    int node = edge_index[(q >> 1) * E + eg];
    const float* feat = scalars + (long long)node * 256 + (q & 1) * 128;
    unsigned short* dst = &sA[el * SA_STRIDE + q * 128];
#pragma unroll
    for (int j = 0; j < 128; j += 8) {
      float4 x0 = *(const float4*)(feat + j);
      float4 x1 = *(const float4*)(feat + j + 4);
      us4 p0 = { f2bf(x0.x), f2bf(x0.y), f2bf(x0.z), f2bf(x0.w) };
      us4 p1 = { f2bf(x1.x), f2bf(x1.y), f2bf(x1.z), f2bf(x1.w) };
      *(us4*)(dst + j)     = p0;
      *(us4*)(dst + j + 4) = p1;
    }
    if (tid < MBLK) {
      int eg2 = ebase + tid;
      sEl[tid] = edge_len[eg2 < E ? eg2 : E - 1];
    }
  }
  __syncthreads();

  const int lane = tid & 31;
  const int ln   = lane & 15;     // A-row / B-col within tile
  const int half = lane >> 4;     // lane half selects K sub-block
  const int n0   = (tid >> 5) * 32;

  v8f acc[4][2];
#pragma unroll
  for (int m = 0; m < 4; ++m)
#pragma unroll
    for (int t = 0; t < 2; ++t) {
      v8f z = {0.f, 0.f, 0.f, 0.f, 0.f, 0.f, 0.f, 0.f};
      acc[m][t] = z;
    }

  // ---- layer 1: [64x512] x [512x256], K in 16 chunks of 32 ----
#pragma unroll 4
  for (int kc = 0; kc < 16; ++kc) {
    const int kbA = kc * 32 + half * 16;  // A layout: 16 contiguous K per lane-half
    v16bf a[4];
#pragma unroll
    for (int m = 0; m < 4; ++m) {
      const unsigned short* p = &sA[(m * 16 + ln) * SA_STRIDE + kbA];
      a[m] = mk_frag(*(const us8*)p, *(const us8*)(p + 8));
    }
    const int kbB = kc * 32 + half * 16;  // B layout: 16 contiguous K per lane-half
    v16bf b[2];
#pragma unroll
    for (int t = 0; t < 2; ++t)
      b[t] = frag16(W1p + (n0 + t * 16 + ln) * 512 + kbB);
    if (kc + 1 < 16)  // CDNA5 global_prefetch_b8 for next weight chunk
      __builtin_prefetch(W1p + (n0 + ln) * 512 + (kc + 1) * 32, 0, 1);
#pragma unroll
    for (int m = 0; m < 4; ++m)
#pragma unroll
      for (int t = 0; t < 2; ++t)
        acc[m][t] = __builtin_amdgcn_wmma_f32_16x16x32_bf16(
            false, a[m], false, b[t], (short)0, acc[m][t], false, false);
  }

  // ---- epilogue 1: + b1 + edge_len * W1[:,512], SiLU, -> bf16 LDS ----
#pragma unroll
  for (int t = 0; t < 2; ++t) {
    const int c = n0 + t * 16 + ln;
    const float bias = b1[c];
    const float w512 = W1[c * 513 + 512];
#pragma unroll
    for (int m = 0; m < 4; ++m)
#pragma unroll
      for (int r = 0; r < 8; ++r) {
        const int e = m * 16 + half * 8 + r;   // D layout: row = r + 8*half
        float x = acc[m][t][r] + bias + sEl[e] * w512;
        float s = x / (1.0f + __expf(-x));     // SiLU
        sH[e * SH_STRIDE + c] = f2bf(s);
      }
  }
  __syncthreads();

  // ---- layer 2: [64x256] x [256x256], K in 8 chunks of 32 ----
  v8f acc2[4][2];
#pragma unroll
  for (int m = 0; m < 4; ++m)
#pragma unroll
    for (int t = 0; t < 2; ++t) {
      v8f z = {0.f, 0.f, 0.f, 0.f, 0.f, 0.f, 0.f, 0.f};
      acc2[m][t] = z;
    }
#pragma unroll 4
  for (int kc = 0; kc < 8; ++kc) {
    const int kbA = kc * 32 + half * 16;
    v16bf a[4];
#pragma unroll
    for (int m = 0; m < 4; ++m) {
      const unsigned short* p = &sH[(m * 16 + ln) * SH_STRIDE + kbA];
      a[m] = mk_frag(*(const us8*)p, *(const us8*)(p + 8));
    }
    const int kbB = kc * 32 + half * 16;
    v16bf b[2];
#pragma unroll
    for (int t = 0; t < 2; ++t)
      b[t] = frag16(W2p + (n0 + t * 16 + ln) * 256 + kbB);
#pragma unroll
    for (int m = 0; m < 4; ++m)
#pragma unroll
      for (int t = 0; t < 2; ++t)
        acc2[m][t] = __builtin_amdgcn_wmma_f32_16x16x32_bf16(
            false, a[m], false, b[t], (short)0, acc2[m][t], false, false);
  }

  // ---- epilogue 2: + b2, f32 store ----
#pragma unroll
  for (int t = 0; t < 2; ++t) {
    const int c = n0 + t * 16 + ln;
    const float bias = b2[c];
#pragma unroll
    for (int m = 0; m < 4; ++m)
#pragma unroll
      for (int r = 0; r < 8; ++r) {
        const int e  = m * 16 + half * 8 + r;
        const int eg = ebase + e;
        if (eg < E) out[(long long)eg * 256 + c] = acc2[m][t][r] + bias;
      }
  }
}

extern "C" void kernel_launch(void* const* d_in, const int* in_sizes, int n_in,
                              void* d_out, int out_size, void* d_ws, size_t ws_size,
                              hipStream_t stream) {
  const float* scalars    = (const float*)d_in[0];
  const int*   edge_index = (const int*)d_in[1];
  const float* edge_len   = (const float*)d_in[2];
  const float* W1         = (const float*)d_in[3];
  const float* b1         = (const float*)d_in[4];
  const float* W2         = (const float*)d_in[5];
  const float* b2         = (const float*)d_in[6];
  float* out = (float*)d_out;
  const int E = in_sizes[2];

  unsigned short* W1p = (unsigned short*)d_ws;              // 256*512 bf16
  unsigned short* W2p = W1p + 256 * 512;                    // 256*256 bf16

  const int pack_elems = 256 * 512 + 256 * 256;
  pack_weights_kernel<<<(pack_elems + 255) / 256, 256, 0, stream>>>(W1, W2, W1p, W2p);

  const int nblk = (E + MBLK - 1) / MBLK;
  edge_mlp_kernel<<<nblk, 256, 0, stream>>>(scalars, edge_index, edge_len,
                                            W1, b1, b2, W1p, W2p, out, E);
}
